// GA_CSWinTransformer_72421738545901
// MI455X (gfx1250) — matrix operand order repaired and checked by reading, not compiled, not run on hardware
//
#include <hip/hip_runtime.h>
#include <hip/hip_bf16.h>
#include <math.h>

typedef unsigned short u16;
typedef __attribute__((ext_vector_type(16))) __bf16 bf16x16;
typedef __attribute__((ext_vector_type(8)))  float  f32x8;
typedef int gv4i __attribute__((vector_size(16)));   // matches builtin param

#define DIM   128
#define RESO  56
#define LSEQ  (RESO*RESO)          // 3136
#define BATCH 32
#define MTOT  (BATCH*LSEQ)         // 100352

#define AS1 __attribute__((address_space(1)))
#define AS3 __attribute__((address_space(3)))

#if defined(__has_builtin)
# if __has_builtin(__builtin_amdgcn_global_load_async_to_lds_b128)
#  define ASYNC_STAGE 1
# endif
#endif
#ifndef ASYNC_STAGE
# define ASYNC_STAGE 0
#endif

__device__ __forceinline__ u16 f2bf(float f) {
  union { float f; unsigned u; } v; v.f = f;
  unsigned r = v.u + 0x7FFFu + ((v.u >> 16) & 1u);
  return (u16)(r >> 16);
}
__device__ __forceinline__ float bf2f(u16 h) {
  union { unsigned u; float f; } v; v.u = ((unsigned)h) << 16;
  return v.f;
}

union AF { bf16x16 v; uint4 q[2]; u16 h[16]; };
union CF { f32x8  v; float f[8]; };

template <int N>
__device__ __forceinline__ void wait_async_le() {
#if ASYNC_STAGE
# if __has_builtin(__builtin_amdgcn_s_wait_asynccnt)
  __builtin_amdgcn_s_wait_asynccnt(N);
# else
  asm volatile("s_wait_asynccnt %0" :: "i"(N) : "memory");
# endif
#endif
}

// ---------------------------------------------------------------------------
// Pack f32 weight [K,N] row-major into CDNA5 WMMA B-fragment tiles (bf16).
// Tile t = kt*(N/16)+nt holds 512 u16: [lane 0..31][elem 0..15],
//   elem e of lane l = W[kt*32 + (l<16 ? e : 16+e)][nt*16 + (l&15)]
// ---------------------------------------------------------------------------
__global__ void pack_w_bf16(const float* __restrict__ W, u16* __restrict__ out,
                            int K, int N) {
  int idx = blockIdx.x * 256 + threadIdx.x;
  if (idx >= K * N) return;
  int e = idx & 15, l = (idx >> 4) & 31, t = idx >> 9;
  int ntn = N >> 4;
  int kt = t / ntn, nt = t - kt * ntn;
  int kk = kt * 32 + ((l < 16) ? e : 16 + e);
  int nn = nt * 16 + (l & 15);
  out[idx] = f2bf(W[(size_t)kk * N + nn]);
}

// ---------------------------------------------------------------------------
// LayerNorm over C=128, one wave32 per row, output bf16.
// ---------------------------------------------------------------------------
__global__ __launch_bounds__(256) void ln_bf16(const float* __restrict__ x,
                                               const float* __restrict__ g,
                                               const float* __restrict__ b,
                                               u16* __restrict__ out) {
  int row  = blockIdx.x * 8 + (threadIdx.x >> 5);
  int lane = threadIdx.x & 31;
  const float4 vv = ((const float4*)(x + (size_t)row * DIM))[lane];
  float s = vv.x + vv.y + vv.z + vv.w;
  #pragma unroll
  for (int o = 16; o > 0; o >>= 1) s += __shfl_xor(s, o, 32);
  float mu = s * (1.0f / DIM);
  float dx = vv.x - mu, dy = vv.y - mu, dz = vv.z - mu, dw = vv.w - mu;
  float q = dx*dx + dy*dy + dz*dz + dw*dw;
  #pragma unroll
  for (int o = 16; o > 0; o >>= 1) q += __shfl_xor(q, o, 32);
  float rs = rsqrtf(q * (1.0f / DIM) + 1e-5f);
  int c = lane * 4;
  u16* po = out + (size_t)row * DIM + c;
  po[0] = f2bf(dx * rs * g[c+0] + b[c+0]);
  po[1] = f2bf(dy * rs * g[c+1] + b[c+1]);
  po[2] = f2bf(dz * rs * g[c+2] + b[c+2]);
  po[3] = f2bf(dw * rs * g[c+3] + b[c+3]);
}

// ---------------------------------------------------------------------------
// bf16 WMMA GEMM: out[M,N] = A[M,K] @ W[K,N], A bf16 row-major,
// Bp = pre-packed fragment tiles. Block = 8 waves; wave tile 32x64 (8 WMMAs
// per K-step). B panel double-buffered in LDS; staged via async global->LDS
// DMA (ASYNCcnt) when the toolchain exposes it, sync copy otherwise.
// epi: 0 = plain -> bf16 | 2 = +bias,GELU -> bf16 | 3 = +bias,+resid -> f32
// ---------------------------------------------------------------------------
__global__ __launch_bounds__(256) void gemm_bf16_wmma(
    const u16* __restrict__ A, const u16* __restrict__ Bp,
    const float* __restrict__ bias, const float* __restrict__ resid,
    u16* __restrict__ outb, float* __restrict__ outf,
    int K, int N, int epi) {
  __shared__ u16 lb[2][2048];                   // 2 x (32K x 64N) packed panel
  int tid = threadIdx.x, lane = tid & 31, wave = tid >> 5;
  int m0 = blockIdx.x * 256 + wave * 32;
  int n0 = blockIdx.y * 64;
  int ntn = N >> 4, nkt = K >> 5;
  int rsel = lane & 15, hsel = lane >> 4;

  CF acc[8];
  #pragma unroll
  for (int j = 0; j < 8; ++j)
    #pragma unroll
    for (int r = 0; r < 8; ++r) acc[j].f[r] = 0.0f;

  auto issue = [&](int kt, int buf) {
    const u16* src = Bp + ((size_t)kt * ntn + (n0 >> 4)) * 512 + tid * 8;
    u16* dst = &lb[buf][tid * 8];
#if ASYNC_STAGE
    __builtin_amdgcn_global_load_async_to_lds_b128(
        (AS1 gv4i*)src, (AS3 gv4i*)dst, 0, 0);
#else
    *(uint4*)dst = *(const uint4*)src;
#endif
  };

  issue(0, 0);
  for (int kt = 0; kt < nkt; ++kt) {
    int buf = kt & 1;
    if (kt + 1 < nkt) {                         // overlap DMA of next panel
      issue(kt + 1, buf ^ 1);
      wait_async_le<1>();                       // panel kt landed in LDS
    } else {
      wait_async_le<0>();
    }
    __syncthreads();                            // whole panel visible

    AF a0, a1;
    const uint4* pa0 = (const uint4*)(A + (size_t)(m0 + rsel) * K + kt * 32 + hsel * 16);
    a0.q[0] = pa0[0]; a0.q[1] = pa0[1];
    const uint4* pa1 = (const uint4*)(A + (size_t)(m0 + 16 + rsel) * K + kt * 32 + hsel * 16);
    a1.q[0] = pa1[0]; a1.q[1] = pa1[1];
    if (kt + 1 < nkt)
      __builtin_prefetch(A + (size_t)(m0 + rsel) * K + kt * 32 + 32, 0, 3);

    #pragma unroll
    for (int j = 0; j < 4; ++j) {
      AF bfr;
      const uint4* pb = (const uint4*)(&lb[buf][j * 512 + lane * 16]);
      bfr.q[0] = pb[0]; bfr.q[1] = pb[1];
      acc[j].v = __builtin_amdgcn_wmma_f32_16x16x32_bf16(
          false, a0.v, false, bfr.v, (short)0, acc[j].v, false, false);
      acc[4 + j].v = __builtin_amdgcn_wmma_f32_16x16x32_bf16(
          false, a1.v, false, bfr.v, (short)0, acc[4 + j].v, false, false);
    }
    __syncthreads();                            // done reading this buffer
  }

  #pragma unroll
  for (int s = 0; s < 2; ++s) {
    #pragma unroll
    for (int j = 0; j < 4; ++j) {
      int col  = n0 + j * 16 + rsel;
      int row0 = m0 + s * 16 + hsel * 8;
      float bv = (epi != 0) ? bias[col] : 0.0f;
      #pragma unroll
      for (int r = 0; r < 8; ++r) {
        size_t off = (size_t)(row0 + r) * N + col;
        float v = acc[s * 4 + j].f[r] + bv;
        if (epi == 2) v = 0.5f * v * (1.0f + erff(v * 0.70710678118f));
        if (epi == 3) outf[off] = v + resid[off];
        else          outb[off] = f2bf(v);
      }
    }
  }
}

// ---------------------------------------------------------------------------
// CSWin window attention + fused LePE, one block per (branch, window, head).
// tok = 112, hd = 32.  Everything in LDS; QK^T and PV on WMMA bf16.
// ---------------------------------------------------------------------------
__global__ __launch_bounds__(256) void cswin_attn(
    const u16* __restrict__ qkv,   // [M, 384] bf16 (q|k|v)
    const float* __restrict__ cw0, const float* __restrict__ cb0,
    const float* __restrict__ cw1, const float* __restrict__ cb1,
    u16* __restrict__ out)         // [M, 128] bf16
{
  extern __shared__ char smem[];
  u16*  qs   = (u16*)smem;                 // 112 x 32, pre-scaled q
  u16*  ks   = qs  + 112 * 32;             // 112 x 32
  u16*  vsT  = ks  + 112 * 32;             // 32 x 128 (transposed, K padded)
  u16*  Ps   = vsT + 32 * 128;             // 112 x 128 (softmax probs, padded)
  float* Ss  = (float*)(Ps + 112 * 128);   // 112 x 112 scores
  float* lep = Ss + 112 * 112;             // 112 x 32 LePE

  int bid = blockIdx.x;
  int br  = bid / 1792; int rem = bid - br * 1792;
  int win = rem >> 1,  head = rem & 1;
  int b   = win / 28,  wi   = win - b * 28;
  const float* cw = br ? cw1 : cw0;
  const float* cb = br ? cb1 : cb0;
  int cbase = head * 32;                   // channel within 64-wide branch
  int qoff  = br * 64 + cbase;             // channel within C=128
  size_t rowbase = (size_t)b * LSEQ;
  int tid = threadIdx.x, lane = tid & 31, wave = tid >> 5;
  int rsel = lane & 15, hsel = lane >> 4;
  const float scale = 0.17677669529f;      // 32^-0.5

  auto tok2pix = [&](int t, int& y, int& x) {
    if (br == 0) { y = t >> 1;            x = wi * 2 + (t & 1); }           // 56x2
    else         { int yq = t / RESO; y = wi * 2 + yq; x = t - yq * RESO; } // 2x56
  };

  // ---- stage q (scaled), k, v(transposed) into LDS -----------------------
  for (int i = tid; i < 448; i += 256) {   // 112 tokens x 4 chunks of 8 ch
    int t = i >> 2, part = i & 3;
    int y, x; tok2pix(t, y, x);
    size_t grow = (rowbase + (size_t)y * RESO + x) * 384;

    uint4 vq = *(const uint4*)(qkv + grow + qoff + part * 8);
    const u16* hq = (const u16*)&vq;
    alignas(16) u16 oq[8];
    #pragma unroll
    for (int jj = 0; jj < 8; ++jj) oq[jj] = f2bf(bf2f(hq[jj]) * scale);
    *(uint4*)(qs + t * 32 + part * 8) = *(const uint4*)oq;

    *(uint4*)(ks + t * 32 + part * 8) =
        *(const uint4*)(qkv + grow + 128 + qoff + part * 8);

    uint4 vv = *(const uint4*)(qkv + grow + 256 + qoff + part * 8);
    const u16* hv = (const u16*)&vv;
    #pragma unroll
    for (int jj = 0; jj < 8; ++jj) vsT[(part * 8 + jj) * 128 + t] = hv[jj];
  }
  for (int i = tid; i < 32 * 16; i += 256)   // zero K-pad tokens 112..127
    vsT[(i >> 4) * 128 + 112 + (i & 15)] = 0;
  __syncthreads();

  // ---- S = q @ k^T : 7x7 tiles of 16x16, K = 32 (one WMMA each) ----------
  for (int tile = wave; tile < 49; tile += 8) {
    int mi = tile / 7, ni = tile - mi * 7;
    AF a, bfr;
    const uint4* pa = (const uint4*)(qs + (mi * 16 + rsel) * 32 + hsel * 16);
    a.q[0] = pa[0]; a.q[1] = pa[1];
    const uint4* pb = (const uint4*)(ks + (ni * 16 + rsel) * 32 + hsel * 16);
    bfr.q[0] = pb[0]; bfr.q[1] = pb[1];
    CF c;
    #pragma unroll
    for (int r = 0; r < 8; ++r) c.f[r] = 0.0f;
    c.v = __builtin_amdgcn_wmma_f32_16x16x32_bf16(
        false, a.v, false, bfr.v, (short)0, c.v, false, false);
    float* so = Ss + (mi * 16 + hsel * 8) * 112 + ni * 16 + rsel;
    #pragma unroll
    for (int r = 0; r < 8; ++r) so[r * 112] = c.f[r];
  }
  __syncthreads();

  // ---- softmax rows (threads 0..111); LePE conv (threads 112..255) -------
  if (tid < 112) {
    float* srow = Ss + tid * 112;
    float mx = -1e30f;
    for (int j = 0; j < 112; ++j) mx = fmaxf(mx, srow[j]);
    float sum = 0.0f;
    for (int j = 0; j < 112; ++j) { float e = __expf(srow[j] - mx); srow[j] = e; sum += e; }
    float inv = 1.0f / sum;
    u16* prow = Ps + tid * 128;
    for (int j = 0; j < 112; ++j)   prow[j] = f2bf(srow[j] * inv);
    for (int j = 112; j < 128; ++j) prow[j] = 0;
  } else {
    int Hsp = br ? 2 : RESO, Wsp = br ? RESO : 2;
    for (int i = tid - 112; i < 3584; i += 144) {
      int t = i >> 5, d = i & 31;
      int yq  = (br == 0) ? (t >> 1) : (t / RESO);
      int ysp = yq;
      int xsp = (br == 0) ? (t & 1) : (t - yq * RESO);
      int c = cbase + d;
      float acc = cb[c];
      #pragma unroll
      for (int dy = -1; dy <= 1; ++dy) {
        int yy = ysp + dy; if (yy < 0 || yy >= Hsp) continue;
        #pragma unroll
        for (int dxo = -1; dxo <= 1; ++dxo) {
          int xx = xsp + dxo; if (xx < 0 || xx >= Wsp) continue;
          acc += cw[c * 9 + (dy + 1) * 3 + (dxo + 1)]
               * bf2f(vsT[d * 128 + yy * Wsp + xx]);
        }
      }
      lep[t * 32 + d] = acc;
    }
  }
  __syncthreads();

  // ---- O = P @ v + lepe : 7x2 tiles, K = 128 (4 WMMAs each) --------------
  for (int tile = wave; tile < 14; tile += 8) {
    int mi = tile >> 1, ni = tile & 1;
    CF c;
    #pragma unroll
    for (int r = 0; r < 8; ++r) c.f[r] = 0.0f;
    #pragma unroll
    for (int k0 = 0; k0 < 128; k0 += 32) {
      AF a, bfr;
      const uint4* pa = (const uint4*)(Ps + (mi * 16 + rsel) * 128 + k0 + hsel * 16);
      a.q[0] = pa[0]; a.q[1] = pa[1];
      const uint4* pb = (const uint4*)(vsT + (ni * 16 + rsel) * 128 + k0 + hsel * 16);
      bfr.q[0] = pb[0]; bfr.q[1] = pb[1];
      c.v = __builtin_amdgcn_wmma_f32_16x16x32_bf16(
          false, a.v, false, bfr.v, (short)0, c.v, false, false);
    }
    int d0 = ni * 16 + rsel;
    #pragma unroll
    for (int r = 0; r < 8; ++r) {
      int t = mi * 16 + hsel * 8 + r;
      int y, x; tok2pix(t, y, x);
      float v = c.f[r] + lep[t * 32 + d0];
      out[(rowbase + (size_t)y * RESO + x) * DIM + qoff + d0] = f2bf(v);
    }
  }
}

// ---------------------------------------------------------------------------
extern "C" void kernel_launch(void* const* d_in, const int* in_sizes, int n_in,
                              void* d_out, int out_size, void* d_ws, size_t ws_size,
                              hipStream_t stream) {
  (void)in_sizes; (void)n_in; (void)out_size; (void)ws_size;
  const float* x      = (const float*)d_in[0];
  const float* n1g    = (const float*)d_in[1];
  const float* n1b    = (const float*)d_in[2];
  const float* qkv_w  = (const float*)d_in[3];
  const float* proj_w = (const float*)d_in[4];
  const float* proj_b = (const float*)d_in[5];
  const float* cw0    = (const float*)d_in[6];
  const float* cb0    = (const float*)d_in[7];
  const float* cw1    = (const float*)d_in[8];
  const float* cb1    = (const float*)d_in[9];
  const float* n2g    = (const float*)d_in[10];
  const float* n2b    = (const float*)d_in[11];
  const float* fc1_w  = (const float*)d_in[12];
  const float* fc1_b  = (const float*)d_in[13];
  const float* fc2_w  = (const float*)d_in[14];
  const float* fc2_b  = (const float*)d_in[15];
  float* out = (float*)d_out;

  char* ws = (char*)d_ws;
  const size_t M = MTOT;
  // region [0, M*512*2): qkv (M*384*2) + xln (M*128*2); later reused as fc1 out
  u16*   qkvB = (u16*)(ws);
  u16*   xln  = (u16*)(ws + M * 384 * 2);
  u16*   hB   = (u16*)(ws);                 // fc1 output (reuse, qkv+xln dead)
  size_t off  = M * 512 * 2;
  u16*   attn = (u16*)(ws + off); off += M * 128 * 2;
  float* x2   = (float*)(ws + off); off += M * 128 * 4;
  u16*   yB   = (u16*)(ws + off); off += M * 128 * 2;
  u16*   wqP  = (u16*)(ws + off); off += 128 * 384 * 2;
  u16*   wpP  = (u16*)(ws + off); off += 128 * 128 * 2;
  u16*   w1P  = (u16*)(ws + off); off += 128 * 512 * 2;
  u16*   w2P  = (u16*)(ws + off); off += 512 * 128 * 2;

  // 1) pack weights to bf16 WMMA fragment layout
  pack_w_bf16<<<(128*384 + 255)/256, 256, 0, stream>>>(qkv_w, wqP, 128, 384);
  pack_w_bf16<<<(128*128 + 255)/256, 256, 0, stream>>>(proj_w, wpP, 128, 128);
  pack_w_bf16<<<(128*512 + 255)/256, 256, 0, stream>>>(fc1_w, w1P, 128, 512);
  pack_w_bf16<<<(512*128 + 255)/256, 256, 0, stream>>>(fc2_w, w2P, 512, 128);

  // 2) LN1 -> bf16
  ln_bf16<<<M / 8, 256, 0, stream>>>(x, n1g, n1b, xln);

  // 3) QKV GEMM [M,128]x[128,384] -> bf16
  gemm_bf16_wmma<<<dim3(M/256, 6), 256, 0, stream>>>(
      xln, wqP, nullptr, nullptr, qkvB, nullptr, 128, 384, 0);

  // 4) CSWin window attention + LePE (2 branches x 896 windows x 2 heads)
  const int ATTN_SMEM = (112*32 + 112*32 + 32*128 + 112*128) * 2
                      + (112*112 + 112*32) * 4;   // 115,712 B
  cswin_attn<<<3584, 256, ATTN_SMEM, stream>>>(qkvB, cw0, cb0, cw1, cb1, attn);

  // 5) proj GEMM + bias + residual(x) -> x2 (f32)
  gemm_bf16_wmma<<<dim3(M/256, 2), 256, 0, stream>>>(
      attn, wpP, proj_b, x, nullptr, x2, 128, 128, 3);

  // 6) LN2 -> bf16
  ln_bf16<<<M / 8, 256, 0, stream>>>(x2, n2g, n2b, yB);

  // 7) FC1 GEMM + bias + exact GELU -> bf16
  gemm_bf16_wmma<<<dim3(M/256, 8), 256, 0, stream>>>(
      yB, w1P, fc1_b, nullptr, hB, nullptr, 128, 512, 2);

  // 8) FC2 GEMM + bias + residual(x2) -> f32 output
  gemm_bf16_wmma<<<dim3(M/256, 2), 256, 0, stream>>>(
      hB, w2P, fc2_b, x2, nullptr, out, 512, 128, 3);
}
